// MultiLevelFeatureSampler_48945447305232
// MI455X (gfx1250) — compile-verified, matrix operand order checked by reading, not run on hardware
//
#include <hip/hip_runtime.h>

typedef __bf16 bf16_t;
typedef __attribute__((ext_vector_type(16))) __bf16 v16bf;
typedef __attribute__((ext_vector_type(8)))  __bf16 v8bf;
typedef __attribute__((ext_vector_type(8)))  float  v8f;

#define NPTS    2048
#define NCH     256
#define NOUT    256
#define KTOT    83
#define KPAD    96      // 3 WMMA K-steps of 32
#define LDSK    104     // padded LDS row stride (bf16 elems): 208B, kills bank conflicts
#define M_WG    32      // points per workgroup
#define THREADS 256     // 8 wave32s

// Assemble a 16-bit A/B WMMA fragment from an LDS row.
// CDNA5 16-bit A 16x32 layout: lane half (lane>=16) selects K+8;
// VGPR0..3 hold K = base..base+7 (contiguous), VGPR4..7 hold K = base+16..base+23.
// -> two 128-bit LDS loads per fragment.
__device__ __forceinline__ v16bf load_frag(const bf16_t* __restrict__ rowp, int s, int half) {
    const int kb = s * 32 + half * 8;
    v8bf lo = *(const v8bf*)(rowp + kb);
    v8bf hi = *(const v8bf*)(rowp + kb + 16);
    v16bf f;
#pragma unroll
    for (int e = 0; e < 8; ++e) { f[e] = lo[e]; f[e + 8] = hi[e]; }
    return f;
}

__global__ __launch_bounds__(THREADS, 1)
void sampler_fused_gemm(const float* __restrict__ points,
                        const float* __restrict__ feat1,
                        const float* __restrict__ feat2,
                        const float* __restrict__ feat3,
                        const float* __restrict__ fc_w,
                        const float* __restrict__ fc_b,
                        float* __restrict__ out)
{
    __shared__ __align__(16) bf16_t Wlds[NOUT * LDSK];  // 53,248 B, bf16 weights K-padded
    __shared__ __align__(16) bf16_t Alds[M_WG * LDSK];  //  6,656 B, gathered patch tile
    __shared__ float pxy[M_WG * 2];

    const int tid   = threadIdx.x;
    const int ch    = blockIdx.y;          // channel (batched-GEMM batch index)
    const int nbase = blockIdx.x * M_WG;   // first point of this block

    // ---- stage points (contiguous 64 floats) ----
    if (tid < M_WG * 2) pxy[tid] = points[(size_t)nbase * 2 + tid];

    // ---- stage weights as bf16, K padded 83 -> 96 with zeros ----
    for (int t = tid; t < NOUT * KPAD; t += THREADS) {
        const int o  = t / KPAD;
        const int kk = t - o * KPAD;
        const float v = (kk < KTOT) ? fc_w[o * KTOT + kk] : 0.0f;
        Wlds[o * LDSK + kk] = (bf16_t)v;
    }
    __syncthreads();

    // ---- gather A tile: X[m, i] = feat_level(i)[ch, oy, ox], bf16 into LDS ----
    for (int t = tid; t < M_WG * KPAD; t += THREADS) {
        const int m = t / KPAD;
        const int i = t - m * KPAD;
        float v = 0.0f;
        if (i < KTOT) {
            int j, k, D;
            const float* fp;
            if (i < 49)      { j = i;      k = 7; D = 128; fp = feat1; }
            else if (i < 74) { j = i - 49; k = 5; D = 64;  fp = feat2; }
            else             { j = i - 74; k = 3; D = 32;  fp = feat3; }
            const int   half = k >> 1;
            const float fD   = (float)(D - 1);
            const float px = pxy[m * 2 + 0];
            const float py = pxy[m * 2 + 1];
            const float x  = fminf(fmaxf(px * fD, 0.0f), fD);
            const float y  = fminf(fmaxf(py * fD, 0.0f), fD);
            // off_x fast-varying (j%k), off_y slow-varying (j/k), clip then truncate
            const float oxf = fminf(fmaxf(x + (float)(j % k - half), 0.0f), fD);
            const float oyf = fminf(fmaxf(y + (float)(j / k - half), 0.0f), fD);
            const int ix = (int)oxf;
            const int iy = (int)oyf;
            v = fp[(size_t)ch * D * D + iy * D + ix];
        }
        Alds[m * LDSK + i] = (bf16_t)v;
    }
    __syncthreads();

    // ---- WMMA: each wave computes a 16(M) x 64(N) slab ----
    const int lane = tid & 31;
    const int wave = tid >> 5;
    const int mt   = wave & 1;    // M-tile within the 32-point block
    const int ng   = wave >> 1;   // N group of 64 outputs
    const int half = lane >> 4;
    const int lrow = lane & 15;

    const bf16_t* arow = &Alds[(mt * 16 + lrow) * LDSK];
    const v16bf a0 = load_frag(arow, 0, half);
    const v16bf a1 = load_frag(arow, 1, half);
    const v16bf a2 = load_frag(arow, 2, half);

#pragma unroll
    for (int t = 0; t < 4; ++t) {
        const int o = ng * 64 + t * 16 + lrow;      // output column (N)
        const bf16_t* brow = &Wlds[o * LDSK];
        const v16bf b0 = load_frag(brow, 0, half);
        const v16bf b1 = load_frag(brow, 1, half);
        const v16bf b2 = load_frag(brow, 2, half);

        // init accumulator with bias: each lane's 8 C VGPRs all belong to column o
        const float bias = fc_b[o];
        v8f acc = {bias, bias, bias, bias, bias, bias, bias, bias};

        acc = __builtin_amdgcn_wmma_f32_16x16x32_bf16(false, a0, false, b0, (short)0, acc, false, false);
        acc = __builtin_amdgcn_wmma_f32_16x16x32_bf16(false, a1, false, b1, (short)0, acc, false, false);
        acc = __builtin_amdgcn_wmma_f32_16x16x32_bf16(false, a2, false, b2, (short)0, acc, false, false);

        // C layout: VGPR r -> M = r + 8*half, N = lane&15
#pragma unroll
        for (int r = 0; r < 8; ++r) {
            const int n = nbase + mt * 16 + half * 8 + r;
            out[((size_t)ch * NPTS + n) * NOUT + o] = acc[r];
        }
    }
}

extern "C" void kernel_launch(void* const* d_in, const int* in_sizes, int n_in,
                              void* d_out, int out_size, void* d_ws, size_t ws_size,
                              hipStream_t stream) {
    (void)in_sizes; (void)n_in; (void)d_ws; (void)ws_size; (void)out_size;
    const float* points = (const float*)d_in[0];
    const float* feat1  = (const float*)d_in[1];
    const float* feat2  = (const float*)d_in[2];
    const float* feat3  = (const float*)d_in[3];
    const float* fc_w   = (const float*)d_in[4];
    const float* fc_b   = (const float*)d_in[5];
    float* out = (float*)d_out;

    dim3 grid(NPTS / M_WG, NCH);   // 64 x 256 workgroups
    dim3 block(THREADS);
    sampler_fused_gemm<<<grid, block, 0, stream>>>(points, feat1, feat2, feat3,
                                                   fc_w, fc_b, out);
}
